// S2SModel_81819126989024
// MI455X (gfx1250) — matrix operand order repaired and checked by reading, not compiled, not run on hardware
//
#include <hip/hip_runtime.h>

// ---------------------------------------------------------------------------
// Types for CDNA5 WMMA (wave32)
// ---------------------------------------------------------------------------
typedef __attribute__((ext_vector_type(16))) __bf16         v16bf;
typedef __attribute__((ext_vector_type(8)))  float          v8f;
typedef __attribute__((ext_vector_type(16))) unsigned short v16us;

__device__ __forceinline__ unsigned short f2bf(float f) {
    unsigned u = __builtin_bit_cast(unsigned, f);
    u += 0x7FFFu + ((u >> 16) & 1u);   // round-to-nearest-even
    return (unsigned short)(u >> 16);
}

__device__ __forceinline__ float sigf(float x) { return 1.f / (1.f + expf(-x)); }

__device__ __forceinline__ v8f wmma_bf16(v16bf a, v16bf b, v8f c) {
    return __builtin_amdgcn_wmma_f32_16x16x32_bf16(false, a, false, b, (short)0, c, false, false);
}

// A-fragment: 16x32 bf16, row-major. p = A + (m_base + lane%16)*lda + (lane/16)*8.
// elements 0..7 = K k0+h*8..+7 ; elements 8..15 = K k0+16+h*8..+7.
__device__ __forceinline__ v16bf load_a_frag(const unsigned short* p) {
    union { uint4 q[2]; v16us v; } t;
    t.q[0] = *(const uint4*)p;
    t.q[1] = *(const uint4*)(p + 16);
    return __builtin_bit_cast(v16bf, t.v);
}

// B-fragment from packed weights. Packed layout: PB[((k/32)*N + col)*32 + (k%32)].
// Per lane: p = PB + (kc*N + col)*32 + hi*8 -> elements 0..7 = K kc*32+hi*8+0..7,
// elements 8..15 = K kc*32+16+hi*8+0..7 = two b128 loads (mirrors the A-fragment path).
__device__ __forceinline__ v16bf load_pk_frag(const unsigned short* p) {
    union { uint4 q[2]; v16us v; } t;
    t.q[0] = *(const uint4*)p;
    t.q[1] = *(const uint4*)(p + 16);
    return __builtin_bit_cast(v16bf, t.v);
}

// Pack f32 row-major W[K][N] -> bf16 PB[(k/32)*N + col][32]. grid=(ceil(N/256), K).
__global__ void pack_kernel(const float* __restrict__ W, unsigned short* __restrict__ PB, int N) {
    int k = blockIdx.y;
    int col = blockIdx.x * 256 + threadIdx.x;
    if (col >= N) return;
    PB[((size_t)(k >> 5) * N + col) * 32 + (k & 31)] = f2bf(W[(size_t)k * N + col]);
}

// ---------------------------------------------------------------------------
// Generic GEMM: C[M,N](f32 and/or bf16) = A[M,K](bf16) * PB(packed bf16) + bias
// block = 256 threads (8 waves); block tile 16(M) x 128(N); M multiple of 16.
// OOB columns are clamped (results discarded), keeping the K-loop branch-free.
// ---------------------------------------------------------------------------
__global__ void gemm_kernel(const unsigned short* __restrict__ A, int lda,
                            const unsigned short* __restrict__ PB,
                            const float* __restrict__ bias,
                            float* __restrict__ Cf, unsigned short* __restrict__ Cb, int ldc,
                            int N, int K) {
    int lane = threadIdx.x & 31, wave = threadIdx.x >> 5, hi = lane >> 4;
    int m_base = blockIdx.y * 16;
    int col = blockIdx.x * 128 + wave * 16 + (lane & 15);
    bool inb = col < N;
    int colc = inb ? col : (N - 1);

    const unsigned short* pa = A + (size_t)(m_base + (lane & 15)) * lda + hi * 8;
    const unsigned short* wp = PB + (size_t)colc * 32 + hi * 8;
    size_t wstep = (size_t)N * 32;

    v8f acc = (v8f)0.0f;
#pragma unroll 2
    for (int k0 = 0; k0 < K; k0 += 32) {
        v16bf a = load_a_frag(pa);
        v16bf b = load_pk_frag(wp);
        acc = wmma_bf16(a, b, acc);
        pa += 32;
        wp += wstep;
    }
    if (inb) {
        float bs = bias[col];
        if (Cf) {
#pragma unroll
            for (int r = 0; r < 8; ++r)
                Cf[(size_t)(m_base + r + hi * 8) * ldc + col] = acc[r] + bs;
        }
        if (Cb) {
#pragma unroll
            for (int r = 0; r < 8; ++r)
                Cb[(size_t)(m_base + r + hi * 8) * ldc + col] = f2bf(acc[r] + bs);
        }
    }
}

// ---------------------------------------------------------------------------
// Persistent LSTM recurrence (one workgroup, 1024 threads = 32 waves).
// Per step: G(32x1024) = h(32x256) @ er(256x1024, packed bf16) via WMMA
// (4 tiles/wave, tile loop kept rolled to bound VGPR pressure), then gates.
// Dynamic LDS: h_bf16 16KB | c f32 32KB | gate staging 128KB = 176KB.
// ---------------------------------------------------------------------------
__global__ void lstm_seq_kernel(const unsigned short* __restrict__ per,  // packed 256x1024
                                const float* __restrict__ xk,
                                int steps, int reverse,
                                float* __restrict__ enc, unsigned short* __restrict__ enc_bf, int enc_off,
                                float* __restrict__ c_all, unsigned short* __restrict__ c_all_bf) {
    extern __shared__ char smem[];
    unsigned short* h_bf = (unsigned short*)smem;            // [32][256]
    float* cst = (float*)(smem + 16384);                     // [32][256]
    float* gst = (float*)(smem + 16384 + 32768);             // [32][1024]
    int tid = threadIdx.x;
    for (int i = tid; i < 32 * 256; i += 1024) { h_bf[i] = 0; cst[i] = 0.f; }
    __syncthreads();
    int lane = tid & 31, wave = tid >> 5, hi = lane >> 4;

    for (int s = 0; s < steps; ++s) {
        int p = reverse ? (steps - 1 - s) : s;
        // ---- GEMM phase: 128 16x16 tiles (2 M x 64 N), 4 per wave ----
#pragma unroll 1
        for (int ti = 0; ti < 4; ++ti) {
            int tile = wave * 4 + ti;
            int m_base = (tile & 1) * 16;
            int n_base = (tile >> 1) * 16;
            int coln = n_base + (lane & 15);
            const unsigned short* pa = h_bf + (size_t)(m_base + (lane & 15)) * 256 + hi * 8;
            const unsigned short* wp = per + (size_t)coln * 32 + hi * 8;
            v8f acc = (v8f)0.0f;
#pragma unroll 2
            for (int kk = 0; kk < 8; ++kk) {
                v16bf a = load_a_frag(pa);
                v16bf b = load_pk_frag(wp);
                acc = wmma_bf16(a, b, acc);
                pa += 32;
                wp += (size_t)1024 * 32;
            }
#pragma unroll
            for (int r = 0; r < 8; ++r)
                gst[(m_base + r + hi * 8) * 1024 + coln] = acc[r];
        }
        __syncthreads();
        // ---- elementwise gates + state update ----
        for (int i = tid; i < 32 * 256; i += 1024) {
            int m = i >> 8, j = i & 255;
            const float* xr = xk + (size_t)(p * 32 + m) * 1024;
            float gi = gst[m * 1024 + j]       + xr[j];
            float gf = gst[m * 1024 + 256 + j] + xr[256 + j];
            float gg = gst[m * 1024 + 512 + j] + xr[512 + j];
            float go = gst[m * 1024 + 768 + j] + xr[768 + j];
            float cc = sigf(gf) * cst[i] + sigf(gi) * tanhf(gg);
            float hh = sigf(go) * tanhf(cc);
            cst[i] = cc;
            h_bf[i] = f2bf(hh);
            if (enc) {
                size_t idx = ((size_t)m * 256 + p) * 512 + enc_off + j;
                enc[idx] = hh;
                enc_bf[idx] = f2bf(hh);
            }
            if (c_all) {
                size_t idx = ((size_t)s * 32 + m) * 256 + j;
                c_all[idx] = cc;
                c_all_bf[idx] = f2bf(cc);
            }
        }
        __syncthreads();
    }
}

// ---------------------------------------------------------------------------
// Attention + combined build. One block per (t,b) row; 256 threads = S.
// ---------------------------------------------------------------------------
__global__ void attn_kernel(const float* __restrict__ henc, const float* __restrict__ sdec,
                            const float* __restrict__ vatt, const float* __restrict__ enc,
                            const float* __restrict__ c_all, unsigned short* __restrict__ combined) {
    __shared__ float sc[256];
    __shared__ float red[256];
    int row = blockIdx.x;        // row = t*32 + b
    int b = row & 31;
    int s = threadIdx.x;
    const float* hp = henc + (size_t)(b * 256 + s) * 128;
    const float* sd = sdec + (size_t)row * 128;
    float acc = 0.f;
#pragma unroll 4
    for (int e = 0; e < 128; ++e) acc += tanhf(hp[e] + sd[e]) * vatt[e];
    red[s] = acc;
    __syncthreads();
    for (int off = 128; off > 0; off >>= 1) {
        if (s < off) red[s] = fmaxf(red[s], red[s + off]);
        __syncthreads();
    }
    float mx = red[0];
    __syncthreads();
    float ee = expf(acc - mx);
    sc[s] = ee;
    red[s] = ee;
    __syncthreads();
    for (int off = 128; off > 0; off >>= 1) {
        if (s < off) red[s] += red[s + off];
        __syncthreads();
    }
    float inv = 1.f / red[0];
    __syncthreads();
    for (int col = threadIdx.x; col < 512; col += 256) {
        float cs = 0.f;
        for (int ss = 0; ss < 256; ++ss)
            cs += sc[ss] * enc[(size_t)(b * 256 + ss) * 512 + col];
        combined[(size_t)row * 768 + col] = f2bf(cs * inv);
    }
    combined[(size_t)row * 768 + 512 + threadIdx.x] = f2bf(c_all[(size_t)row * 256 + threadIdx.x]);
}

// ---------------------------------------------------------------------------
// Vocab projection + streaming log-softmax stats (packed-lv_w fast path).
// Block = 8 waves; tile 128(M) x 128(N); one B-fragment reused across 8
// M-tiles (immediate-offset A loads from a single base; A padded to 1024 rows,
// pad-tile accumulators are discarded). Packed lv_w streams from HBM -> issue
// global_prefetch several K-chunks ahead. Per-row sum(exp) via LDS atomics
// then one global atomicAdd per row per block; target logit stored directly.
// ---------------------------------------------------------------------------
__global__ void vocab_pk_kernel(const unsigned short* __restrict__ A,   // 1024 x 4096 bf16 (992 valid)
                                const unsigned short* __restrict__ PB,  // packed 4096 x 50000
                                const float* __restrict__ lv_b,
                                const int* __restrict__ targ,
                                float* __restrict__ sumexp, float* __restrict__ logit_tgt) {
    __shared__ float rowsum[128];
    int lane = threadIdx.x & 31, wave = threadIdx.x >> 5, hi = lane >> 4;
    int m0 = blockIdx.y * 128;
    int col = blockIdx.x * 128 + wave * 16 + (lane & 15);
    bool inb = col < 50000;
    int colc = inb ? col : 49999;

    const unsigned short* wp = PB + (size_t)colc * 32 + hi * 8;
    const unsigned short* pa = A + (size_t)(m0 + (lane & 15)) * 4096 + hi * 8;

    v8f acc[8];
#pragma unroll
    for (int i = 0; i < 8; ++i) acc[i] = (v8f)0.0f;

#pragma unroll 1
    for (int k0 = 0; k0 < 4096; k0 += 32) {
        v16bf bfrag = load_pk_frag(wp);
        if (k0 + 128 < 4096) {
            __builtin_prefetch(wp + (size_t)4 * 50000 * 32, 0, 1);  // lv_w: HBM stream, 4 chunks ahead
            __builtin_prefetch(pa + 128, 0, 1);                     // A panel: L2-resident
        }
#pragma unroll
        for (int mt = 0; mt < 8; ++mt) {
            v16bf afrag = load_a_frag(pa + (size_t)mt * 16 * 4096);  // immediate offsets
            acc[mt] = wmma_bf16(afrag, bfrag, acc[mt]);
        }
        pa += 32;
        wp += (size_t)50000 * 32;
    }

    for (int i = threadIdx.x; i < 128; i += 256) rowsum[i] = 0.f;
    __syncthreads();
    float bs = inb ? lv_b[col] : 0.f;
#pragma unroll
    for (int mt = 0; mt < 8; ++mt) {
        int mb = m0 + mt * 16;
        if (mb >= 992 || !inb) continue;
#pragma unroll
        for (int r = 0; r < 8; ++r) {
            int rowL = mt * 16 + r + hi * 8;
            int row = m0 + rowL;
            float l = acc[mt][r] + bs;
            atomicAdd(&rowsum[rowL], expf(l));   // logits are tiny: exp is safe
            int bb = row & 31, t = row >> 5;
            if (col == targ[bb * 32 + t + 1]) logit_tgt[row] = l;
        }
    }
    __syncthreads();
    for (int i = threadIdx.x; i < 128; i += 256) {
        int row = m0 + i;
        if (row < 992) atomicAdd(&sumexp[row], rowsum[i]);
    }
}

// Fallback vocab path when ws can't hold packed lv_w: on-the-fly f32->bf16.
__device__ __forceinline__ v16bf load_b_f32(const float* p) {
    v16us u;
#pragma unroll
    for (int v = 0; v < 8; ++v) {
        u[2 * v]     = f2bf(p[(size_t)(2 * v + (v >= 4 ? 8 : 0)) * 50000]);
        u[2 * v + 1] = f2bf(p[(size_t)(2 * v + 1 + (v >= 4 ? 8 : 0)) * 50000]);
    }
    return __builtin_bit_cast(v16bf, u);
}

__global__ void vocab_f32_kernel(const unsigned short* __restrict__ A,  // 1024 x 4096 (992 valid)
                                 const float* __restrict__ W,           // 4096 x 50000 f32
                                 const float* __restrict__ lv_b,
                                 const int* __restrict__ targ,
                                 float* __restrict__ sumexp, float* __restrict__ logit_tgt) {
    __shared__ float rowsum[128];
    int lane = threadIdx.x & 31, wave = threadIdx.x >> 5, hi = lane >> 4;
    int m0 = blockIdx.y * 128;
    int col = blockIdx.x * 128 + wave * 16 + (lane & 15);
    bool inb = col < 50000;
    int colc = inb ? col : 49999;

    const float* wp = W + (size_t)(hi * 8) * 50000 + colc;
    const unsigned short* pa = A + (size_t)(m0 + (lane & 15)) * 4096 + hi * 8;

    v8f acc[8];
#pragma unroll
    for (int i = 0; i < 8; ++i) acc[i] = (v8f)0.0f;

#pragma unroll 1
    for (int k0 = 0; k0 < 4096; k0 += 32) {
        v16bf bfrag = load_b_f32(wp);
        if (k0 + 96 < 4096)
            __builtin_prefetch(wp + (size_t)96 * 50000, 0, 1);
#pragma unroll
        for (int mt = 0; mt < 8; ++mt) {
            v16bf afrag = load_a_frag(pa + (size_t)mt * 16 * 4096);
            acc[mt] = wmma_bf16(afrag, bfrag, acc[mt]);
        }
        pa += 32;
        wp += (size_t)32 * 50000;
    }

    for (int i = threadIdx.x; i < 128; i += 256) rowsum[i] = 0.f;
    __syncthreads();
    float bs = inb ? lv_b[col] : 0.f;
#pragma unroll
    for (int mt = 0; mt < 8; ++mt) {
        int mb = m0 + mt * 16;
        if (mb >= 992 || !inb) continue;
#pragma unroll
        for (int r = 0; r < 8; ++r) {
            int rowL = mt * 16 + r + hi * 8;
            int row = m0 + rowL;
            float l = acc[mt][r] + bs;
            atomicAdd(&rowsum[rowL], expf(l));
            int bb = row & 31, t = row >> 5;
            if (col == targ[bb * 32 + t + 1]) logit_tgt[row] = l;
        }
    }
    __syncthreads();
    for (int i = threadIdx.x; i < 128; i += 256) {
        int row = m0 + i;
        if (row < 992) atomicAdd(&sumexp[row], rowsum[i]);
    }
}

// ---------------------------------------------------------------------------
// Embedding gathers (f32 -> bf16 rows laid out for WMMA A-matrices)
// ---------------------------------------------------------------------------
__global__ void embed_enc_kernel(const float* __restrict__ emb, const int* __restrict__ inp,
                                 unsigned short* __restrict__ X) {
    int idx = blockIdx.x * 256 + threadIdx.x;       // 8192 rows x 128
    if (idx >= 8192 * 128) return;
    int row = idx >> 7, e = idx & 127;
    int s = row >> 5, b = row & 31;                 // row = s*32 + b
    X[idx] = f2bf(emb[(size_t)inp[b * 256 + s] * 128 + e]);
}

__global__ void embed_dec_kernel(const float* __restrict__ emb, const int* __restrict__ targ,
                                 unsigned short* __restrict__ X) {
    int idx = blockIdx.x * 256 + threadIdx.x;       // 992 rows x 128
    if (idx >= 992 * 128) return;
    int row = idx >> 7, e = idx & 127;
    int t = row >> 5, b = row & 31;                 // row = t*32 + b
    int tok = (t == 0) ? 2 : targ[b * 32 + t];      // START=2, tok_in[:,t]=targ[:,t] (t>=1)
    X[idx] = f2bf(emb[(size_t)tok * 128 + e]);
}

__global__ void init_kernel(float* __restrict__ sumexp, float* __restrict__ logit_tgt) {
    int i = blockIdx.x * 256 + threadIdx.x;
    if (i < 992) { sumexp[i] = 0.f; logit_tgt[i] = 0.f; }
}

// loss = sum_{t,b} mask * (log(sumexp) - logit_tgt) / (B*T) = /1024
__global__ void loss_kernel(const float* __restrict__ sumexp, const float* __restrict__ logit_tgt,
                            const int* __restrict__ targ, float* __restrict__ out) {
    __shared__ float red[1024];
    int i = threadIdx.x;
    float v = 0.f;
    if (i < 992) {
        int b = i & 31, t = i >> 5;
        int tok = targ[b * 32 + t + 1];
        if (tok != 0) v = logf(sumexp[i]) - logit_tgt[i];
    }
    red[i] = v;
    __syncthreads();
    for (int off = 512; off > 0; off >>= 1) {
        if (i < off) red[i] += red[i + off];
        __syncthreads();
    }
    if (i == 0) out[0] = red[0] / 1024.0f;
}

// ---------------------------------------------------------------------------
// Host launcher
// ---------------------------------------------------------------------------
extern "C" void kernel_launch(void* const* d_in, const int* in_sizes, int n_in,
                              void* d_out, int out_size, void* d_ws, size_t ws_size,
                              hipStream_t stream) {
    (void)in_sizes; (void)n_in; (void)out_size;
    const int*   inp    = (const int*)d_in[0];
    const int*   targ   = (const int*)d_in[1];
    const float* emb    = (const float*)d_in[2];
    const float* ek_f   = (const float*)d_in[3];
    const float* er_f   = (const float*)d_in[4];
    const float* eb_f   = (const float*)d_in[5];
    const float* ek_b   = (const float*)d_in[6];
    const float* er_b   = (const float*)d_in[7];
    const float* eb_b   = (const float*)d_in[8];
    const float* dk     = (const float*)d_in[9];
    const float* dr     = (const float*)d_in[10];
    const float* db     = (const float*)d_in[11];
    const float* henc_w = (const float*)d_in[12];
    const float* henc_b = (const float*)d_in[13];
    const float* sdec_w = (const float*)d_in[14];
    const float* sdec_b = (const float*)d_in[15];
    const float* vatt   = (const float*)d_in[16];
    const float* lf_w   = (const float*)d_in[17];
    const float* lf_b   = (const float*)d_in[18];
    const float* lv_w   = (const float*)d_in[19];
    const float* lv_b   = (const float*)d_in[20];

    char* wsb = (char*)d_ws;
    size_t off = 0;
    auto alloc = [&](size_t bytes) -> char* {
        char* p = wsb + off;
        off = (off + bytes + 255) & ~(size_t)255;
        return p;
    };
    unsigned short* Xenc = (unsigned short*)alloc((size_t)8192 * 128 * 2);
    unsigned short* Xdec = (unsigned short*)alloc((size_t)992 * 128 * 2);
    float* xkf  = (float*)alloc((size_t)8192 * 1024 * 4);
    float* xkb  = (float*)alloc((size_t)8192 * 1024 * 4);
    float* dxk  = (float*)alloc((size_t)992 * 1024 * 4);
    float* enc  = (float*)alloc((size_t)8192 * 512 * 4);
    unsigned short* encb = (unsigned short*)alloc((size_t)8192 * 512 * 2);
    float* henc = (float*)alloc((size_t)8192 * 128 * 4);
    float* call = (float*)alloc((size_t)992 * 256 * 4);
    unsigned short* callb = (unsigned short*)alloc((size_t)992 * 256 * 2);
    float* sdec = (float*)alloc((size_t)992 * 128 * 4);
    unsigned short* comb = (unsigned short*)alloc((size_t)992 * 768 * 2);
    unsigned short* ffb  = (unsigned short*)alloc((size_t)1024 * 4096 * 2);  // padded to 1024 rows
    float* sumexp = (float*)alloc((size_t)992 * 4);
    float* ltgt   = (float*)alloc((size_t)992 * 4);
    // packed bf16 weights
    unsigned short* Pekf  = (unsigned short*)alloc((size_t)128 * 1024 * 2);
    unsigned short* Pekb  = (unsigned short*)alloc((size_t)128 * 1024 * 2);
    unsigned short* Pdk   = (unsigned short*)alloc((size_t)128 * 1024 * 2);
    unsigned short* Perf  = (unsigned short*)alloc((size_t)256 * 1024 * 2);
    unsigned short* Perb  = (unsigned short*)alloc((size_t)256 * 1024 * 2);
    unsigned short* Pdr   = (unsigned short*)alloc((size_t)256 * 1024 * 2);
    unsigned short* Phenc = (unsigned short*)alloc((size_t)512 * 128 * 2);
    unsigned short* Psdec = (unsigned short*)alloc((size_t)256 * 128 * 2);
    unsigned short* Plf   = (unsigned short*)alloc((size_t)768 * 4096 * 2);
    size_t base_end = off;
    bool pk_lv = (ws_size >= base_end + (size_t)4096 * 50000 * 2);
    unsigned short* Plv = pk_lv ? (unsigned short*)alloc((size_t)4096 * 50000 * 2) : nullptr;

    auto pack = [&](const float* W, unsigned short* P, int K, int N) {
        pack_kernel<<<dim3((N + 255) / 256, K), 256, 0, stream>>>(W, P, N);
    };

    const size_t lstm_lds = 16384 + 32768 + 131072;  // 176 KB dynamic LDS

    init_kernel<<<4, 256, 0, stream>>>(sumexp, ltgt);
    embed_enc_kernel<<<(8192 * 128) / 256, 256, 0, stream>>>(emb, inp, Xenc);
    embed_dec_kernel<<<(992 * 128 + 255) / 256, 256, 0, stream>>>(emb, targ, Xdec);

    pack(ek_f, Pekf, 128, 1024);
    pack(ek_b, Pekb, 128, 1024);
    pack(dk,   Pdk,  128, 1024);
    pack(er_f, Perf, 256, 1024);
    pack(er_b, Perb, 256, 1024);
    pack(dr,   Pdr,  256, 1024);
    pack(henc_w, Phenc, 512, 128);
    pack(sdec_w, Psdec, 256, 128);
    pack(lf_w,   Plf,   768, 4096);
    if (pk_lv) pack(lv_w, Plv, 4096, 50000);

    // x@k + b for every timestep (batched WMMA GEMMs)
    gemm_kernel<<<dim3(8, 512), 256, 0, stream>>>(Xenc, 128, Pekf, eb_f, xkf, nullptr, 1024, 1024, 128);
    gemm_kernel<<<dim3(8, 512), 256, 0, stream>>>(Xenc, 128, Pekb, eb_b, xkb, nullptr, 1024, 1024, 128);
    gemm_kernel<<<dim3(8, 62),  256, 0, stream>>>(Xdec, 128, Pdk,  db,   dxk, nullptr, 1024, 1024, 128);

    // encoder recurrences (fwd / bwd)
    lstm_seq_kernel<<<1, 1024, lstm_lds, stream>>>(Perf, xkf, 256, 0, enc, encb, 0,   nullptr, nullptr);
    lstm_seq_kernel<<<1, 1024, lstm_lds, stream>>>(Perb, xkb, 256, 1, enc, encb, 256, nullptr, nullptr);

    // henc_proj = enc @ henc_w + b
    gemm_kernel<<<dim3(1, 512), 256, 0, stream>>>(encb, 512, Phenc, henc_b, henc, nullptr, 128, 128, 512);

    // decoder recurrence (teacher-forced: 31 steps), collects all c_t
    lstm_seq_kernel<<<1, 1024, lstm_lds, stream>>>(Pdr, dxk, 31, 0, nullptr, nullptr, 0, call, callb);

    // sdec for all steps = C_all @ sdec_w + b
    gemm_kernel<<<dim3(1, 62), 256, 0, stream>>>(callb, 256, Psdec, sdec_b, sdec, nullptr, 128, 128, 256);

    // attention + combined for all 992 (t,b) rows
    attn_kernel<<<992, 256, 0, stream>>>(henc, sdec, vatt, enc, call, comb);

    // FF: combined @ lf_w + b -> bf16
    gemm_kernel<<<dim3(32, 62), 256, 0, stream>>>(comb, 768, Plf, lf_b, nullptr, ffb, 4096, 4096, 768);

    // vocab projection + streaming log-softmax stats
    if (pk_lv)
        vocab_pk_kernel<<<dim3(391, 8), 256, 0, stream>>>(ffb, Plv, lv_b, targ, sumexp, ltgt);
    else
        vocab_f32_kernel<<<dim3(391, 8), 256, 0, stream>>>(ffb, lv_w, lv_b, targ, sumexp, ltgt);

    loss_kernel<<<1, 1024, 0, stream>>>(sumexp, ltgt, targ, (float*)d_out);
}
